// Emu3VQVAEVectorQuantizer_20469814133517
// MI455X (gfx1250) — matrix operand order repaired and checked by reading, not compiled, not run on hardware
//
#include <hip/hip_runtime.h>

// ---------------------------------------------------------------------------
// Emu3 VQ-VAE vector quantizer for gfx1250 (MI455X).
//   hidden : (8,1,256,32,32) f32   codebook : (16384,256) f32
//   out    : (8,1,32,32) argmin indices, written as float
// Core: 8192x16384x256 GEMM on v_wmma_f32_16x16x32_bf16 + fused argmin,
// codebook tiles staged block-wide via the Tensor Data Mover into
// double-buffered LDS (s_wait_tensorcnt + barriers). K is split 4 ways
// across blocks (grid=256) to cover all WGPs; a combine pass merges.
// ---------------------------------------------------------------------------

#define NQ   8192      // number of query vectors (B*T*H*W)
#define KCB  16384     // codebook entries
#define CDIM 256       // embedding dim

#define WAVES    8             // waves per block (256 threads)
#define TILE_K   32            // codebook rows staged per iteration
#define KSPLIT   4             // K segments (one per block column)
#define SEG_K    (KCB / KSPLIT)
#define SEG_TILES (SEG_K / TILE_K)     // 128 iterations per block
#define LDS_ROW  528           // 512B data + 16B TDM pad -> bank-spread rows
#define LDS_TILE (TILE_K * LDS_ROW)   // 16896 B per buffer

typedef __attribute__((ext_vector_type(16))) __bf16          v16bf;
typedef __attribute__((ext_vector_type(8)))  unsigned short  v8u;
typedef __attribute__((ext_vector_type(16))) unsigned short  v16u;
typedef __attribute__((ext_vector_type(8)))  float           v8f;
typedef __attribute__((ext_vector_type(4)))  float           v4f;
typedef __attribute__((ext_vector_type(4)))  unsigned int    v4u32;
typedef __attribute__((ext_vector_type(8)))  int             v8i32;
typedef __attribute__((ext_vector_type(4)))  int             v4i32;

#if defined(__has_builtin)
#if __has_builtin(__builtin_amdgcn_tensor_load_to_lds) && \
    __has_builtin(__builtin_amdgcn_s_wait_tensorcnt)
#define HAVE_TDM 1
#endif
#endif
#ifndef HAVE_TDM
#define HAVE_TDM 0
#endif

__device__ __forceinline__ unsigned short f2bf_rne(float f) {
    unsigned int u = __float_as_uint(f);
    u += 0x7fffu + ((u >> 16) & 1u);      // round-to-nearest-even
    return (unsigned short)(u >> 16);
}

// --- Kernel 1: (B,T,C,H,W) f32 -> (N=B*H*W, C) bf16, coalesced over H*W ----
__global__ void vq_xpose_cast(const float* __restrict__ hs,
                              unsigned short* __restrict__ xb) {
    const int bid = blockIdx.x;            // 0 .. 8*256-1
    const int b   = bid >> 8;              // batch
    const int c   = bid & 255;             // channel
    const float* src = hs + ((size_t)(b * CDIM + c)) * 1024;  // H*W = 1024
    for (int i = threadIdx.x; i < 1024; i += blockDim.x) {
        xb[((size_t)(b * 1024 + i)) * CDIM + c] = f2bf_rne(src[i]);
    }
}

// --- Kernel 2: codebook f32 -> bf16, plus e_sq[k] = sum(e^2) in f32 --------
__global__ void vq_codebook_prep(const float* __restrict__ cbf,
                                 unsigned short* __restrict__ cbb,
                                 float* __restrict__ esq) {
    const int gtid = blockIdx.x * blockDim.x + threadIdx.x;
    const int row  = gtid >> 5;            // one wave32 per codebook row
    const int lane = threadIdx.x & 31;
    const float* r = cbf + (size_t)row * CDIM + lane * 8;
    v4f a = *reinterpret_cast<const v4f*>(r);
    v4f b = *reinterpret_cast<const v4f*>(r + 4);
    float s = 0.f;
    unsigned short* w = cbb + (size_t)row * CDIM + lane * 8;
#pragma unroll
    for (int i = 0; i < 4; ++i) {
        s += a[i] * a[i] + b[i] * b[i];
        w[i]     = f2bf_rne(a[i]);
        w[i + 4] = f2bf_rne(b[i]);
    }
#pragma unroll
    for (int m = 16; m >= 1; m >>= 1) s += __shfl_xor(s, m, 32);
    if (lane == 0) esq[row] = s;
}

#if HAVE_TDM
// Issue one TDM transfer: 32 codebook rows (32 x 512B) -> LDS at ldsOff,
// 16B padding inserted after every 512B row (LDS row stride = 528B).
// D# layout per CDNA5 ISA ch.8: group0 = {count/lds_addr/global_addr/type},
// group1 = {data_size=8B, pad_en, pad_interval=512B, pad_amount=16B,
//           tensor_dim0=64, tensor_dim1=16384, tile=64x32, dim0_stride=64}.
__device__ __forceinline__ void tdm_load_tile(const unsigned short* cbb,
                                              int k0, unsigned ldsOff) {
    unsigned long long ga =
        (unsigned long long)(const void*)(cbb + (size_t)k0 * CDIM);
    v4u32 g0;
    g0[0] = 1u;                                        // count=1, no gather
    g0[1] = ldsOff;                                    // lds_addr (bytes)
    g0[2] = (unsigned)(ga & 0xffffffffu);              // global_addr[31:0]
    g0[3] = (unsigned)((ga >> 32) & 0x01ffffffu)       // global_addr[56:32]
          | 0x80000000u;                               // type=2 ("image")
    v8i32 g1;
    g1[0] = (int)((3u << 16)      // data_size = 8B
                | (1u << 20)      // pad_enable
                | (6u << 22)      // pad_interval: 128 DWORDs = 512B
                | (3u << 25));    // pad_amount: 4 DWORDs = 16B
    g1[1] = (int)(64u << 16);     // tensor_dim0 = 64 (8B units per row)
    g1[2] = (int)((unsigned)KCB << 16);  // tensor_dim1[15:0] = 16384 rows
    g1[3] = (int)(64u << 16);     // tile_dim0 = 64 (full 512B row)
    g1[4] = TILE_K;               // tile_dim1 = 32 rows
    g1[5] = 64;                   // tensor_dim0_stride = 64 (contiguous rows)
    g1[6] = 0;
    g1[7] = 0;
    v4i32 z4 = {0, 0, 0, 0};      // groups 2/3 unused (2-D tensor)
    v8i32 z8 = {0, 0, 0, 0, 0, 0, 0, 0};
    // clang-23 / therock-10.0 arity: (g0, g1, g2, g3, extra, cpol)
    __builtin_amdgcn_tensor_load_to_lds(g0, g1, z4, z4, z8, 0);
}
#endif

// --- Kernel 3: WMMA GEMM + fused argmin over one K segment -----------------
// blockIdx.x = seg*64 + rowgroup. Block = 8 waves x 16 rows = 128 query
// rows scanning SEG_K codebook entries; per-segment (min,idx) -> workspace.
__global__ __launch_bounds__(256)
void vq_wmma_argmin(const unsigned short* __restrict__ xb,
                    const unsigned short* __restrict__ cbb,
                    const float* __restrict__ esq,
                    float* __restrict__ partv,
                    int*   __restrict__ parti) {
    __shared__ __align__(16) unsigned char ldsTile[2 * LDS_TILE];
    // The TDM writes ldsTile through a descriptor (no pointer arg), so the
    // compiler sees a never-stored array and would fold all LDS reads to
    // undef. Escape the address into opaque asm with a memory clobber so
    // the staged data must actually be loaded (keeps addrspace(3) => ds_*).
    asm volatile("" ::"v"(&ldsTile[0]) : "memory");

    const int lane = threadIdx.x & 31;
    const int half = lane >> 4;            // 0: lanes 0-15, 1: lanes 16-31
    const int m16  = lane & 15;
    const int wave = threadIdx.x >> 5;
    const int seg  = blockIdx.x >> 6;      // K segment 0..3
    const int rg   = blockIdx.x & 63;      // row group
    const int n0   = (rg * WAVES + wave) * 16;   // first query row
    const int kseg = seg * SEG_K;

    union Pun { v16u u; v8u h[2]; v16bf bf; };

    // ---- A operands: lane half h holds channels with ((c>>4)&1)==h --------
    const unsigned short* xrow = xb + (size_t)(n0 + m16) * CDIM;
    v16bf A[8];
#pragma unroll
    for (int w = 0; w < 8; ++w) {
        Pun a;
        a.h[0] = *reinterpret_cast<const v8u*>(xrow + 32 * w + 16 * half);
        a.h[1] = *reinterpret_cast<const v8u*>(xrow + 32 * w + 16 * half + 8);
        A[w] = a.bf;
    }

    float bestv[8];
    int   besti[8];
#pragma unroll
    for (int r = 0; r < 8; ++r) { bestv[r] = 3.0e38f; besti[r] = kseg; }

#if HAVE_TDM
    if (wave == 0) tdm_load_tile(cbb, kseg, 0u);   // prime buffer 0
#endif

    for (int t = 0; t < SEG_TILES; ++t) {
        const int k0 = kseg + t * TILE_K;
        const unsigned char* buf = ldsTile + (unsigned)(t & 1) * LDS_TILE;

#if HAVE_TDM
        if (wave == 0) {
            if (t + 1 < SEG_TILES) {
                // prefetch next tile into the other buffer (safe: trailing
                // barrier of t-1 guaranteed all reads of it finished)
                tdm_load_tile(cbb, k0 + TILE_K,
                              (unsigned)((t + 1) & 1) * LDS_TILE);
                __builtin_amdgcn_s_wait_tensorcnt(1);  // tile `t` landed
            } else {
                __builtin_amdgcn_s_wait_tensorcnt(0);
            }
        }
        __syncthreads();                 // tile `t` visible to all waves
#else
        // Fallback: cooperative global->LDS staging with the same layout.
        {
            unsigned char* dbuf = const_cast<unsigned char*>(buf);
            const unsigned char* src =
                reinterpret_cast<const unsigned char*>(cbb) + ((size_t)k0 << 9);
#pragma unroll
            for (int r = 0; r < 4; ++r) {              // 1024 16B segs / 256 thr
                const int idx = (int)threadIdx.x + 256 * r;
                const int row = idx >> 5, sg = idx & 31;
                v8u d = *reinterpret_cast<const v8u*>(src + ((size_t)row << 9)
                                                          + sg * 16);
                *reinterpret_cast<v8u*>(dbuf + row * LDS_ROW + sg * 16) = d;
            }
        }
        __syncthreads();
#endif
        // TDM (or staging) may have written this buffer: clobber point so
        // the B-operand LDS loads below cannot be folded away or CSE'd
        // across the ping-pong iterations.
        asm volatile("" ::"v"(buf) : "memory");

        const float esj0 = esq[k0 + m16];
        const float esj1 = esq[k0 + 16 + m16];
        const int   j0   = k0 + m16;
        const int   j1   = k0 + 16 + m16;

        // B operand: col = lane&15, channel = 32w + e + 16*half (e = 0..15)
        const unsigned char* b0p = buf + m16 * LDS_ROW + 32 * half;
        const unsigned char* b1p = b0p + 16 * LDS_ROW;

        v8f acc0 = {}, acc1 = {};
#pragma unroll
        for (int w = 0; w < 8; ++w) {
            Pun b0, b1;
            b0.h[0] = *reinterpret_cast<const v8u*>(b0p + 64 * w);
            b0.h[1] = *reinterpret_cast<const v8u*>(b0p + 64 * w + 16);
            b1.h[0] = *reinterpret_cast<const v8u*>(b1p + 64 * w);
            b1.h[1] = *reinterpret_cast<const v8u*>(b1p + 64 * w + 16);
            acc0 = __builtin_amdgcn_wmma_f32_16x16x32_bf16(
                       false, A[w], false, b0.bf, (short)0, acc0, false, false);
            acc1 = __builtin_amdgcn_wmma_f32_16x16x32_bf16(
                       false, A[w], false, b1.bf, (short)0, acc1, false, false);
        }

        // D layout: VGPR r, lane -> (M = r + 8*half, N = lane&15)
#pragma unroll
        for (int r = 0; r < 8; ++r) {
            const float d0 = esj0 - 2.0f * acc0[r];    // |e|^2 - 2 x.e
            if (d0 < bestv[r]) { bestv[r] = d0; besti[r] = j0; }
            const float d1 = esj1 - 2.0f * acc1[r];
            if (d1 < bestv[r]) { bestv[r] = d1; besti[r] = j1; }
        }

        __syncthreads();                 // all reads of tile `t` done
    }

    // ---- Reduce argmin across the 16 lanes sharing each row ---------------
#pragma unroll
    for (int r = 0; r < 8; ++r) {
        float v = bestv[r];
        int   i = besti[r];
#pragma unroll
        for (int m = 1; m < 16; m <<= 1) {
            const float ov = __shfl_xor(v, m, 32);
            const int   oi = __shfl_xor(i, m, 32);
            if (ov < v || (ov == v && oi < i)) { v = ov; i = oi; }
        }
        if (m16 == 0) {
            const int n = n0 + r + 8 * half;
            partv[seg * NQ + n] = v;
            parti[seg * NQ + n] = i;
        }
    }
}

// --- Kernel 4: combine the KSPLIT per-segment candidates per row -----------
// Segments combined in ascending order with strict < keeps the lowest
// index on exact ties (argmin first-occurrence semantics).
__global__ void vq_combine(const float* __restrict__ partv,
                           const int* __restrict__ parti,
                           float* __restrict__ out) {
    const int n = blockIdx.x * blockDim.x + threadIdx.x;
    float bv = partv[n];
    int   bi = parti[n];
#pragma unroll
    for (int s = 1; s < KSPLIT; ++s) {
        const float v = partv[s * NQ + n];
        const int   i = parti[s * NQ + n];
        if (v < bv || (v == bv && i < bi)) { bv = v; bi = i; }
    }
    out[n] = (float)bi;
}

// ---------------------------------------------------------------------------
extern "C" void kernel_launch(void* const* d_in, const int* in_sizes, int n_in,
                              void* d_out, int out_size, void* d_ws, size_t ws_size,
                              hipStream_t stream) {
    const float* hidden   = (const float*)d_in[0];   // (8,1,256,32,32) f32
    const float* codebook = (const float*)d_in[1];   // (16384,256) f32
    float* out = (float*)d_out;                      // 8192 indices as float

    char* ws = (char*)d_ws;
    unsigned short* xb  = (unsigned short*)ws;                      // 4 MiB
    ws += (size_t)NQ * CDIM * sizeof(unsigned short);
    unsigned short* cbb = (unsigned short*)ws;                      // 8 MiB
    ws += (size_t)KCB * CDIM * sizeof(unsigned short);
    float* esq = (float*)ws;                                        // 64 KiB
    ws += (size_t)KCB * sizeof(float);
    float* partv = (float*)ws;                                      // 128 KiB
    ws += (size_t)KSPLIT * NQ * sizeof(float);
    int* parti = (int*)ws;                                          // 128 KiB

    // 1) transpose + cast queries: 8*256 blocks, coalesced over H*W
    vq_xpose_cast<<<dim3(8 * 256), dim3(256), 0, stream>>>(hidden, xb);

    // 2) codebook cast + |e|^2: one wave per row, 8 waves/block
    vq_codebook_prep<<<dim3(KCB / 8), dim3(256), 0, stream>>>(codebook, cbb, esq);

    // 3) TDM-staged WMMA GEMM + argmin: 256 blocks (4 K-segments x 64 row
    //    groups), 8 waves x 16 rows each
    vq_wmma_argmin<<<dim3(KSPLIT * (NQ / (WAVES * 16))), dim3(32 * WAVES), 0,
                     stream>>>(xb, cbb, esq, partv, parti);

    // 4) merge per-segment candidates -> final indices (as float)
    vq_combine<<<dim3(NQ / 256), dim3(256), 0, stream>>>(partv, parti, out);
}
